// GraphAttentionEncoder_18803366822027
// MI455X (gfx1250) — matrix operand order repaired and checked by reading, not compiled, not run on hardware
//
#include <hip/hip_runtime.h>
#include <hip/hip_bf16.h>
#include <cstddef>
#include <cstdint>

// ---------------------------------------------------------------------------
// GraphAttentionEncoder on gfx1250 (MI455X, wave32, WMMA).
// Pipeline: f32->f16 convert, 8 WMMA GEMMs (fused bias + lrelu/relu),
// one distance kernel (reused by both blocks), two fused attention kernels.
// ---------------------------------------------------------------------------

#define DINNER 256     // feature dim D
#define KNBR   32      // neighbors per node
#define BETA_C   0.2f
#define ALPHA_C  0.3f
#define MAXD_C   250.0f

typedef __attribute__((ext_vector_type(16))) _Float16 v16h;
typedef __attribute__((ext_vector_type(8)))  _Float16 v8h;
typedef __attribute__((ext_vector_type(8)))  float    v8f;

union Frag16 { v16h v; v8h h[2]; };

// ---------------------------------------------------------------------------
// f32 -> f16 conversion
// ---------------------------------------------------------------------------
__global__ void cvt_f32_f16(const float* __restrict__ src,
                            _Float16* __restrict__ dst, int n) {
    int i = blockIdx.x * blockDim.x + threadIdx.x;
    if (i < n) dst[i] = (_Float16)src[i];
}

// ---------------------------------------------------------------------------
// WMMA GEMM: C[n,o] = sum_d A[n,d] * W[o,d] + bias[o]   (Din fixed = 256)
// A: f16 [Nrows,256] row-major; W: f16 [Dout,256] row-major.
// EPI: 0 = none, 1 = leaky_relu(0.01), 2 = relu.
// Block = 128 threads = 4 waves; wave -> one 16x16 output tile.
// Grid: x = ceil(Nrows/64), y = Dout/16.
// ---------------------------------------------------------------------------
template <int EPI>
__global__ void gemm_wmma(const _Float16* __restrict__ A,
                          const _Float16* __restrict__ W,
                          const float* __restrict__ bias,
                          float* __restrict__ Cf,
                          _Float16* __restrict__ Ch,
                          int Nrows, int Dout) {
    const int lane  = threadIdx.x & 31;
    const int wave  = threadIdx.x >> 5;
    const int rowBase = (blockIdx.x * 4 + wave) * 16;
    if (rowBase >= Nrows) return;              // wave-uniform guard (N % 16 == 0)
    const int colBase = blockIdx.y * 16;

    const int lhalf = lane >> 4;               // 0 or 1
    const int l16   = lane & 15;

    // A fragment: row = rowBase + l16; halves 0..7 at k = lhalf*8 + c, 8..15 at +16
    const _Float16* Arow = A + (size_t)(rowBase + l16) * DINNER + lhalf * 8;
    // B fragment: weight row (output col) = colBase + l16; halves 0..7 at k = lhalf*8 + c, 8..15 at +16
    const _Float16* Brow = W + (size_t)(colBase + l16) * DINNER + lhalf * 8;

    v8f acc = {};
#pragma unroll
    for (int kb = 0; kb < DINNER; kb += 32) {
        Frag16 a, b;
        a.h[0] = *(const v8h*)(Arow + kb);
        a.h[1] = *(const v8h*)(Arow + kb + 16);
        b.h[0] = *(const v8h*)(Brow + kb);
        b.h[1] = *(const v8h*)(Brow + kb + 16);
        acc = __builtin_amdgcn_wmma_f32_16x16x32_f16(
            /*neg_a=*/false, a.v, /*neg_b=*/false, b.v,
            /*c_mod=*/(short)0, acc, /*reuse_a=*/false, /*reuse_b=*/false);
    }

    const int col = colBase + l16;
    const float bc = bias[col];
#pragma unroll
    for (int v = 0; v < 8; ++v) {
        float r = acc[v] + bc;
        if (EPI == 1) r = (r > 0.f) ? r : 0.01f * r;
        if (EPI == 2) r = fmaxf(r, 0.f);
        const size_t idx = (size_t)(rowBase + lhalf * 8 + v) * Dout + col;
        if (Cf) Cf[idx] = r;
        if (Ch) Ch[idx] = (_Float16)r;
    }
}

// ---------------------------------------------------------------------------
// Distance weights: dw[n,k] = exp(-ALPHA * ||x_n - x_{col[n,k]}|| / (250+1e-8))
// One block (256 threads) per node: 32 groups of 8 lanes, each group one
// neighbor, 8-lane shuffle reduction over 256 channels.
// ---------------------------------------------------------------------------
__global__ void dist_kernel(const float* __restrict__ xF,
                            const int* __restrict__ col,
                            float* __restrict__ dw) {
    __shared__ int cS[KNBR];
    const int n = blockIdx.x;
    if (threadIdx.x < KNBR) cS[threadIdx.x] = col[(size_t)n * KNBR + threadIdx.x];
    __syncthreads();

    const int k = threadIdx.x >> 3;
    const int j = threadIdx.x & 7;
    const float* xn = xF + (size_t)n * DINNER;
    const float* xc = xF + (size_t)cS[k] * DINNER;
    float s = 0.f;
#pragma unroll 8
    for (int i = 0; i < DINNER / 8; ++i) {
        const float t = xn[j + i * 8] - xc[j + i * 8];
        s += t * t;
    }
    s += __shfl_xor(s, 1, 32);
    s += __shfl_xor(s, 2, 32);
    s += __shfl_xor(s, 4, 32);
    if (j == 0)
        dw[(size_t)n * KNBR + k] =
            __expf(-(ALPHA_C / (MAXD_C + 1e-8f)) * sqrtf(s));
}

// ---------------------------------------------------------------------------
// Fused attention: per (node, channel) softmax over 32 neighbors + weighted
// sum + context mix. One block (256 threads = channels) per node.
//   s_k  = yna[col_k, d] * dw[n,k]
//   attw = softmax_k(s)
//   nb   = sum_k attw_k * x[col_k, d]
//   ctx  = BETA * ysa[n,d] * t[n,d] + (1-BETA) * nb   -> f16 (next GEMM input)
// ---------------------------------------------------------------------------
__global__ void attn_kernel(const float* __restrict__ xF,   // neighbor source (x)
                            const float* __restrict__ tF,   // target features
                            const float* __restrict__ ysa,  // target @ Wsa.T + b
                            const float* __restrict__ yna,  // x @ Wna.T + b
                            const float* __restrict__ dw,   // [N,K]
                            const int* __restrict__ col,    // [N,K]
                            _Float16* __restrict__ ctxh) {
    __shared__ int   cS[KNBR];
    __shared__ float wS[KNBR];
    const int n = blockIdx.x;
    if (threadIdx.x < KNBR) {
        cS[threadIdx.x] = col[(size_t)n * KNBR + threadIdx.x];
        wS[threadIdx.x] = dw[(size_t)n * KNBR + threadIdx.x];
    }
    __syncthreads();

    const int d = threadIdx.x;
    float sv[KNBR];
    float mx = -3.402823466e+38f;
#pragma unroll
    for (int k = 0; k < KNBR; ++k) {
        sv[k] = yna[(size_t)cS[k] * DINNER + d] * wS[k];
        mx = fmaxf(mx, sv[k]);
    }
    float denom = 0.f, acc = 0.f;
#pragma unroll
    for (int k = 0; k < KNBR; ++k) {
        const float e = __expf(sv[k] - mx);
        denom += e;
        acc = fmaf(e, xF[(size_t)cS[k] * DINNER + d], acc);
    }
    const float nb  = acc / denom;
    const size_t id = (size_t)n * DINNER + d;
    const float ctx = BETA_C * ysa[id] * tF[id] + (1.0f - BETA_C) * nb;
    ctxh[id] = (_Float16)ctx;
}

// ---------------------------------------------------------------------------
// Host-side launch sequence
// ---------------------------------------------------------------------------
extern "C" void kernel_launch(void* const* d_in, const int* in_sizes, int n_in,
                              void* d_out, int out_size, void* d_ws, size_t ws_size,
                              hipStream_t stream) {
    (void)n_in; (void)out_size; (void)ws_size;

    const float* x  = (const float*)d_in[0];
    const int*   ei = (const int*)d_in[1];
    const float* Wf[8]; const float* bf[8];
    for (int i = 0; i < 8; ++i) { Wf[i] = (const float*)d_in[2 + i]; bf[i] = (const float*)d_in[10 + i]; }
    // Weight order: 0 W_sa1, 1 W_na1, 2 W_f1a, 3 W_f1b, 4 W_sa2, 5 W_na2, 6 W_f2a, 7 W_f2b(64x256)

    const int N = in_sizes[0] / DINNER;      // 10000
    const int E = in_sizes[1] / 2;           // N*K
    const int* col = ei + E;                 // edge_index[1]

    // Workspace carve-out (256B aligned)
    char* ws = (char*)d_ws; size_t off = 0;
    auto carve = [&](size_t bytes) { void* p = ws + off; off = (off + bytes + 255) & ~(size_t)255; return p; };
    _Float16* xh   = (_Float16*)carve((size_t)N * DINNER * 2);
    _Float16* Wh[8];
    for (int i = 0; i < 7; ++i) Wh[i] = (_Float16*)carve((size_t)DINNER * DINNER * 2);
    Wh[7] = (_Float16*)carve((size_t)64 * DINNER * 2);
    float*    bufA = (float*)carve((size_t)N * DINNER * 4);   // y_sa
    float*    bufB = (float*)carve((size_t)N * DINNER * 4);   // y_na
    _Float16* ctxh = (_Float16*)carve((size_t)N * DINNER * 2);
    _Float16* zH   = (_Float16*)carve((size_t)N * DINNER * 2);
    float*    hF   = (float*)carve((size_t)N * DINNER * 4);
    _Float16* hH   = (_Float16*)carve((size_t)N * DINNER * 2);
    float*    dwB  = (float*)carve((size_t)N * KNBR * 4);

    const dim3 blk128(128), blk256(256);
    const dim3 gemmGrid256((N + 63) / 64, DINNER / 16);
    const dim3 gemmGrid64((N + 63) / 64, 64 / 16);

    // 1) convert inputs to f16
    {
        int n = N * DINNER;
        cvt_f32_f16<<<(n + 255) / 256, blk256, 0, stream>>>(x, xh, n);
        for (int i = 0; i < 7; ++i) {
            int wn = DINNER * DINNER;
            cvt_f32_f16<<<(wn + 255) / 256, blk256, 0, stream>>>(Wf[i], Wh[i], wn);
        }
        int wn = 64 * DINNER;
        cvt_f32_f16<<<(wn + 255) / 256, blk256, 0, stream>>>(Wf[7], Wh[7], wn);
    }

    // 2) distance weights (shared by both attention blocks)
    dist_kernel<<<N, blk256, 0, stream>>>(x, col, dwB);

    // ---- Block 1 (target = x) ----
    gemm_wmma<0><<<gemmGrid256, blk128, 0, stream>>>(xh, Wh[0], bf[0], bufA, nullptr, N, DINNER); // y_sa1
    gemm_wmma<0><<<gemmGrid256, blk128, 0, stream>>>(xh, Wh[1], bf[1], bufB, nullptr, N, DINNER); // y_na1
    attn_kernel<<<N, blk256, 0, stream>>>(x, x, bufA, bufB, dwB, col, ctxh);
    gemm_wmma<1><<<gemmGrid256, blk128, 0, stream>>>(ctxh, Wh[2], bf[2], nullptr, zH, N, DINNER); // lrelu
    gemm_wmma<2><<<gemmGrid256, blk128, 0, stream>>>(zH, Wh[3], bf[3], hF, hH, N, DINNER);        // h = relu(...)

    // ---- Block 2 (target = h, neighbors still from x) ----
    gemm_wmma<0><<<gemmGrid256, blk128, 0, stream>>>(hH, Wh[4], bf[4], bufA, nullptr, N, DINNER); // y_sa2
    gemm_wmma<0><<<gemmGrid256, blk128, 0, stream>>>(xh, Wh[5], bf[5], bufB, nullptr, N, DINNER); // y_na2
    attn_kernel<<<N, blk256, 0, stream>>>(x, hF, bufA, bufB, dwB, col, ctxh);
    gemm_wmma<1><<<gemmGrid256, blk128, 0, stream>>>(ctxh, Wh[6], bf[6], nullptr, zH, N, DINNER); // lrelu
    gemm_wmma<2><<<gemmGrid64, blk128, 0, stream>>>(zH, Wh[7], bf[7], (float*)d_out, nullptr, N, 64); // relu out
}
